// DecoderLayer_22136261444120
// MI455X (gfx1250) — matrix operand order repaired and checked by reading, not compiled, not run on hardware
//
#include <hip/hip_runtime.h>
#include <math.h>

// ---------------------------------------------------------------------------
// Types for CDNA5 WMMA (wave32, 16x16x32 bf16 -> f32)
// ---------------------------------------------------------------------------
typedef __bf16 bf16;
typedef __attribute__((ext_vector_type(16))) __bf16 bf16x16;
typedef __attribute__((ext_vector_type(8)))  __bf16 bf16x8;
typedef __attribute__((ext_vector_type(8)))  float  f32x8;

union ABFrag { bf16x16 v; bf16x8 h[2]; };

// A-operand (16x32, MxK): lane m(=lane&15) holds row m; low half-wave holds
// K = k0+0..7 and k0+16..23, high half-wave K = k0+8..15 and k0+24..31.
__device__ __forceinline__ bf16x16 load_frag_a(const bf16* A, int lda, int row0, int k0) {
  int lane = threadIdx.x & 31;
  const bf16* p = A + (size_t)(row0 + (lane & 15)) * lda + k0 + ((lane & 16) >> 1);
  ABFrag u;
  u.h[0] = *(const bf16x8*)(p);
  u.h[1] = *(const bf16x8*)(p + 16);
  return u.v;
}

// B-operand (32x16, KxN): lane n(=lane&15) holds column n; low half-wave
// K = k0+0..7 and k0+16..23, high half-wave K = k0+8..15 and k0+24..31.
// W stored row-major (N x K) -> per-lane loads are two contiguous 16-byte runs.
__device__ __forceinline__ bf16x16 load_frag_b(const bf16* W, int ldw, int col0, int k0) {
  int lane = threadIdx.x & 31;
  const bf16* p = W + (size_t)(col0 + (lane & 15)) * ldw + k0 + ((lane & 16) >> 1);
  ABFrag u;
  u.h[0] = *(const bf16x8*)(p);
  u.h[1] = *(const bf16x8*)(p + 16);
  return u.v;
}

__device__ __forceinline__ f32x8 wmma_bf16(bf16x16 a, bf16x16 b, f32x8 c) {
  return __builtin_amdgcn_wmma_f32_16x16x32_bf16(false, a, false, b, (short)0, c, false, false);
}

// C layout: lane element r -> row r + (lane>=16 ? 8 : 0), col = lane&15.

// ---------------------------------------------------------------------------
// Pack fp32 -> bf16 with K padding (rows x cols -> rows x ldpad, zero pad)
// ---------------------------------------------------------------------------
__global__ void pack_pad_bf16(const float* __restrict__ src, bf16* __restrict__ dst,
                              int rows, int cols, int ldpad) {
  size_t total = (size_t)rows * ldpad;
  for (size_t i = (size_t)blockIdx.x * blockDim.x + threadIdx.x; i < total;
       i += (size_t)gridDim.x * blockDim.x) {
    int r = (int)(i / ldpad);
    int c = (int)(i % ldpad);
    dst[i] = (c < cols) ? (bf16)src[(size_t)r * cols + c] : (bf16)0.0f;
  }
}

// ---------------------------------------------------------------------------
// GEMM: C[M,N] = A[M,Kpad] @ W[N,Kpad]^T + bias, fp32 out
// block = 128 threads (4 waves), block tile 64x64, wave tile 32x32
// ---------------------------------------------------------------------------
__global__ __launch_bounds__(128) void gemm_bf16_f32(
    const bf16* __restrict__ A, int lda,
    const bf16* __restrict__ W, int ldw,
    const float* __restrict__ bias,
    float* __restrict__ C, int ldc, int kIters) {
  int wave = threadIdx.x >> 5;
  int m0 = blockIdx.x * 64 + (wave & 1) * 32;
  int n0 = blockIdx.y * 64 + (wave >> 1) * 32;
  f32x8 acc[2][2] = {};
  for (int kt = 0; kt < kIters; ++kt) {
    int k0 = kt * 32;
    bf16x16 a0 = load_frag_a(A, lda, m0, k0);
    bf16x16 a1 = load_frag_a(A, lda, m0 + 16, k0);
    bf16x16 b0 = load_frag_b(W, ldw, n0, k0);
    bf16x16 b1 = load_frag_b(W, ldw, n0 + 16, k0);
    acc[0][0] = wmma_bf16(a0, b0, acc[0][0]);
    acc[0][1] = wmma_bf16(a0, b1, acc[0][1]);
    acc[1][0] = wmma_bf16(a1, b0, acc[1][0]);
    acc[1][1] = wmma_bf16(a1, b1, acc[1][1]);
  }
  int lane = threadIdx.x & 31;
  int col = lane & 15;
  int rb = (lane & 16) >> 1;
#pragma unroll
  for (int mi = 0; mi < 2; ++mi)
#pragma unroll
    for (int ni = 0; ni < 2; ++ni) {
      int n = n0 + ni * 16 + col;
      float bv = bias[n];
#pragma unroll
      for (int r = 0; r < 8; ++r) {
        int m = m0 + mi * 16 + r + rb;
        C[(size_t)m * ldc + n] = acc[mi][ni][r] + bv;
      }
    }
}

// Same GEMM but relu + bf16 output at column offset 1 (FFN hidden layout)
__global__ __launch_bounds__(128) void gemm_bf16_relu(
    const bf16* __restrict__ A, int lda,
    const bf16* __restrict__ W, int ldw,
    const float* __restrict__ bias,
    bf16* __restrict__ C, int ldc, int col_off, int kIters) {
  int wave = threadIdx.x >> 5;
  int m0 = blockIdx.x * 64 + (wave & 1) * 32;
  int n0 = blockIdx.y * 64 + (wave >> 1) * 32;
  f32x8 acc[2][2] = {};
  for (int kt = 0; kt < kIters; ++kt) {
    int k0 = kt * 32;
    bf16x16 a0 = load_frag_a(A, lda, m0, k0);
    bf16x16 a1 = load_frag_a(A, lda, m0 + 16, k0);
    bf16x16 b0 = load_frag_b(W, ldw, n0, k0);
    bf16x16 b1 = load_frag_b(W, ldw, n0 + 16, k0);
    acc[0][0] = wmma_bf16(a0, b0, acc[0][0]);
    acc[0][1] = wmma_bf16(a0, b1, acc[0][1]);
    acc[1][0] = wmma_bf16(a1, b0, acc[1][0]);
    acc[1][1] = wmma_bf16(a1, b1, acc[1][1]);
  }
  int lane = threadIdx.x & 31;
  int col = lane & 15;
  int rb = (lane & 16) >> 1;
#pragma unroll
  for (int mi = 0; mi < 2; ++mi)
#pragma unroll
    for (int ni = 0; ni < 2; ++ni) {
      int n = n0 + ni * 16 + col;
      float bv = bias[n];
#pragma unroll
      for (int r = 0; r < 8; ++r) {
        int m = m0 + mi * 16 + r + rb;
        C[(size_t)m * ldc + col_off + n] = (bf16)fmaxf(acc[mi][ni][r] + bv, 0.0f);
      }
    }
}

// ---------------------------------------------------------------------------
// QKV packing. sproj: (B*T) x 512 fp32 (spatial). Per head compute Lorentz
// time t = sqrt(1 + sum s^2); pack [sign*t, s(64), zeros->96] per (b,h,t).
// ---------------------------------------------------------------------------
__global__ void qk_pack(const float* __restrict__ sp, bf16* __restrict__ dst, float sign) {
  int idx = blockIdx.x * blockDim.x + threadIdx.x; // (row, h) pairs: 8192*8
  if (idx >= 8192 * 8) return;
  int row = idx >> 3, h = idx & 7;
  int b = row >> 10, t = row & 1023;
  const float* s = sp + (size_t)row * 512 + h * 64;
  float ss = 0.f;
#pragma unroll 8
  for (int d = 0; d < 64; ++d) ss += s[d] * s[d];
  bf16* o = dst + ((size_t)(b * 8 + h) * 1024 + t) * 96;
  o[0] = (bf16)(sign * sqrtf(1.0f + ss));
#pragma unroll 8
  for (int d = 0; d < 64; ++d) o[1 + d] = (bf16)s[d];
  for (int d = 65; d < 96; ++d) o[d] = (bf16)0.0f;
}

// V packed d-major: vtb[(bh*80 + d)*S + t], rows 0..64 = [t_v, s], 65..79 zero.
__global__ void v_pack(const float* __restrict__ sp, bf16* __restrict__ vtb) {
  int bh = blockIdx.y;
  int t = blockIdx.x * blockDim.x + threadIdx.x;
  if (t >= 1024) return;
  int b = bh >> 3, h = bh & 7;
  const float* s = sp + ((size_t)(b * 1024 + t)) * 512 + h * 64;
  float ss = 0.f;
#pragma unroll 8
  for (int d = 0; d < 64; ++d) ss += s[d] * s[d];
  bf16* base = vtb + (size_t)bh * 80 * 1024 + t;
  base[0] = (bf16)sqrtf(1.0f + ss);
#pragma unroll 8
  for (int d = 1; d < 65; ++d) base[(size_t)d * 1024] = (bf16)s[d - 1];
  for (int d = 65; d < 80; ++d) base[(size_t)d * 1024] = (bf16)0.0f;
}

// ---------------------------------------------------------------------------
// Attention: one wave per (b,h, 16-row q tile). Full score row in LDS (fp32),
// softmax stats in lanes, attn@V with on-the-fly exp A-fragments, Lorentz
// projection fused; writes merged spatial output (B*T x 512) bf16.
// ---------------------------------------------------------------------------
__global__ __launch_bounds__(32) void attention_kernel(
    const bf16* __restrict__ qb, const bf16* __restrict__ kb,
    const bf16* __restrict__ vtb, bf16* __restrict__ attn_s,
    int T, int Sdim, int causal) {
  __shared__ float sc[16 * 1024]; // 64 KB
  int bh = blockIdx.x;
  int t0 = blockIdx.y * 16;
  int b = bh >> 3, h = bh & 7;
  const bf16* q = qb + (size_t)bh * T * 96;
  const bf16* k = kb + (size_t)bh * Sdim * 96;
  const bf16* vt = vtb + (size_t)bh * 80 * Sdim;
  int lane = threadIdx.x;
  int col = lane & 15;
  int rb = (lane & 16) >> 1;

  // Pass 1: scores = (2 + 2*(qm . k)) / 8, masked
  for (int s0 = 0; s0 < Sdim; s0 += 16) {
    f32x8 acc = {};
#pragma unroll
    for (int kk = 0; kk < 96; kk += 32) {
      bf16x16 a = load_frag_a(q, 96, t0, kk);
      bf16x16 bfr = load_frag_b(k, 96, s0, kk);
      acc = wmma_bf16(a, bfr, acc);
    }
    int s = s0 + col;
#pragma unroll
    for (int r = 0; r < 8; ++r) {
      int m = r + rb;
      int t = t0 + m;
      float v = (2.0f + 2.0f * acc[r]) * 0.125f;
      if (causal && (s > t)) v = -3.4028235e38f;
      sc[m * 1024 + s] = v;
    }
  }
  __syncthreads();

  // Pass 2: per-row max and 1/sum (lanes 0..15 each own a row)
  float mrow = -3.4028235e38f, isum = 0.f;
  if (lane < 16) {
    const float* row = &sc[lane * 1024];
    for (int s = 0; s < Sdim; ++s) mrow = fmaxf(mrow, row[s]);
    float sum = 0.f;
    for (int s = 0; s < Sdim; ++s) sum += __expf(row[s] - mrow);
    isum = 1.0f / sum;
  }
  float mr = __shfl(mrow, col, 32);
  float ir = __shfl(isum, col, 32);

  // Pass 3: mid = softmax(scores) @ v  (N = 80 padded dims, K = Sdim)
  f32x8 acc[5] = {};
  for (int s0 = 0; s0 < Sdim; s0 += 32) {
    int base = s0 + ((lane & 16) >> 1);
    const float* row = &sc[col * 1024];
    ABFrag u;
#pragma unroll
    for (int j = 0; j < 8; ++j) {
      u.h[0][j] = (bf16)(__expf(row[base + j] - mr) * ir);
      u.h[1][j] = (bf16)(__expf(row[base + 16 + j] - mr) * ir);
    }
#pragma unroll
    for (int nt = 0; nt < 5; ++nt) {
      bf16x16 bfr = load_frag_b(vt, Sdim, nt * 16, s0);
      acc[nt] = wmma_bf16(u.v, bfr, acc[nt]);
    }
  }
  __syncthreads();
  float* mid = sc; // reuse LDS: 16 x 80
#pragma unroll
  for (int nt = 0; nt < 5; ++nt)
#pragma unroll
    for (int r = 0; r < 8; ++r)
      mid[(r + rb) * 80 + nt * 16 + col] = acc[nt][r];
  __syncthreads();

  // Lorentz project + write merged spatial (half a row per lane)
  int r = lane & 15;
  int half = lane >> 4;
  const float* mrp = &mid[r * 80];
  float nrm = mrp[0] * mrp[0];
  for (int d = 1; d < 65; ++d) nrm -= mrp[d] * mrp[d];
  float scale = rsqrtf(fmaxf(nrm, 1e-6f));
  bf16* out = attn_s + ((size_t)(b * T + t0 + r)) * 512 + h * 64;
  for (int d = 0; d < 32; ++d) {
    int dd = 1 + half * 32 + d;
    out[dd - 1] = (bf16)(mrp[dd] * scale);
  }
}

// ---------------------------------------------------------------------------
// Row-wise lresnet + lorentz_ln. One 256-thread block per row.
// hs includes bias. h = [sqrt(1+|hs|^2), hs]; z = x + 3h; project; layernorm
// on spatial; add_time. Writes fp32 (513) and optionally padded bf16 (544).
// ---------------------------------------------------------------------------
__device__ __forceinline__ float block_reduce_sum(float v, float* red) {
  int tid = threadIdx.x;
  red[tid] = v;
  __syncthreads();
  for (int s = 128; s > 0; s >>= 1) {
    if (tid < s) red[tid] += red[tid + s];
    __syncthreads();
  }
  float r = red[0];
  __syncthreads();
  return r;
}

__global__ __launch_bounds__(256) void residual_ln(
    const float* __restrict__ xin, const float* __restrict__ hs,
    const float* __restrict__ g, const float* __restrict__ beta,
    float* __restrict__ xout, bf16* __restrict__ xoutb) {
  __shared__ float red[256];
  int row = blockIdx.x;
  int tid = threadIdx.x;
  const float* h = hs + (size_t)row * 512;
  const float* xi = xin + (size_t)row * 513;

  float a = 0.f;
  for (int i = tid; i < 512; i += 256) a += h[i] * h[i];
  float ht = sqrtf(1.0f + block_reduce_sum(a, red));

  float zs = 0.f;
  for (int i = tid; i < 512; i += 256) {
    float z = xi[1 + i] + 3.0f * h[i];
    zs += z * z;
  }
  float zss = block_reduce_sum(zs, red);
  float z0 = xi[0] + 3.0f * ht;
  float sc = rsqrtf(fmaxf(z0 * z0 - zss, 1e-6f));

  float ms = 0.f;
  for (int i = tid; i < 512; i += 256) ms += (xi[1 + i] + 3.0f * h[i]) * sc;
  float mu = block_reduce_sum(ms, red) * (1.0f / 512.0f);

  float vs = 0.f;
  for (int i = tid; i < 512; i += 256) {
    float s = (xi[1 + i] + 3.0f * h[i]) * sc - mu;
    vs += s * s;
  }
  float var = block_reduce_sum(vs, red) * (1.0f / 512.0f);
  float inv = rsqrtf(var + 1e-5f);

  float ys = 0.f;
  for (int i = tid; i < 512; i += 256) {
    float s = (xi[1 + i] + 3.0f * h[i]) * sc;
    float y = (s - mu) * inv * g[i] + beta[i];
    xout[(size_t)row * 513 + 1 + i] = y;
    if (xoutb) xoutb[(size_t)row * 544 + 1 + i] = (bf16)y;
    ys += y * y;
  }
  float t = sqrtf(1.0f + block_reduce_sum(ys, red));
  if (tid == 0) {
    xout[(size_t)row * 513] = t;
    if (xoutb) xoutb[(size_t)row * 544] = (bf16)t;
  }
  if (xoutb)
    for (int p = 513 + tid; p < 544; p += 256) xoutb[(size_t)row * 544 + p] = (bf16)0.0f;
}

// FFN hidden time: a2 rows [t, relu(2048), pad->2080]; fill t and zero pad.
__global__ __launch_bounds__(256) void ffn_time(bf16* __restrict__ a2) {
  __shared__ float red[256];
  int row = blockIdx.x;
  int tid = threadIdx.x;
  bf16* r = a2 + (size_t)row * 2080;
  float ss = 0.f;
  for (int i = 1 + tid; i <= 2048; i += 256) {
    float v = (float)r[i];
    ss += v * v;
  }
  float tot = block_reduce_sum(ss, red);
  if (tid == 0) r[0] = (bf16)sqrtf(1.0f + tot);
  for (int i = 2049 + tid; i < 2080; i += 256) r[i] = (bf16)0.0f;
}

// ---------------------------------------------------------------------------
// Host launcher
// ---------------------------------------------------------------------------
extern "C" void kernel_launch(void* const* d_in, const int* in_sizes, int n_in,
                              void* d_out, int out_size, void* d_ws, size_t ws_size,
                              hipStream_t stream) {
  (void)in_sizes; (void)n_in; (void)out_size; (void)ws_size;
  const int T = 1024, S = 1024;
  const int M = 8192;      // B*T == B*S
  const int KD = 544;      // 513 padded to 17*32
  const int K2 = 2080;     // 2049 padded to 65*32

  const float* X    = (const float*)d_in[0];
  const float* ENC  = (const float*)d_in[1];
  const float* Wq_s = (const float*)d_in[4];  const float* bq_s = (const float*)d_in[5];
  const float* Wk_s = (const float*)d_in[6];  const float* bk_s = (const float*)d_in[7];
  const float* Wv_s = (const float*)d_in[8];  const float* bv_s = (const float*)d_in[9];
  const float* Wq_c = (const float*)d_in[10]; const float* bq_c = (const float*)d_in[11];
  const float* Wk_c = (const float*)d_in[12]; const float* bk_c = (const float*)d_in[13];
  const float* Wv_c = (const float*)d_in[14]; const float* bv_c = (const float*)d_in[15];
  const float* Wo_s = (const float*)d_in[16]; const float* bo_s = (const float*)d_in[17];
  const float* Wo_c = (const float*)d_in[18]; const float* bo_c = (const float*)d_in[19];
  const float* W1   = (const float*)d_in[20]; const float* b1   = (const float*)d_in[21];
  const float* W2   = (const float*)d_in[22]; const float* b2   = (const float*)d_in[23];
  const float* g1   = (const float*)d_in[24]; const float* be1  = (const float*)d_in[25];
  const float* g2   = (const float*)d_in[26]; const float* be2  = (const float*)d_in[27];
  const float* g3   = (const float*)d_in[28]; const float* be3  = (const float*)d_in[29];

  char* p = (char*)d_ws;
  auto alloc = [&](size_t bytes) -> char* {
    char* r = p;
    p += (bytes + 255) & ~(size_t)255;
    return r;
  };
  bf16*  xb   = (bf16*)alloc((size_t)M * KD * 2);
  bf16*  encb = (bf16*)alloc((size_t)M * KD * 2);
  bf16*  actb = (bf16*)alloc((size_t)M * KD * 2);
  bf16*  qb   = (bf16*)alloc((size_t)64 * T * 96 * 2);
  bf16*  kb   = (bf16*)alloc((size_t)64 * S * 96 * 2);
  bf16*  vtb  = (bf16*)alloc((size_t)64 * 80 * S * 2);
  bf16*  attn = (bf16*)alloc((size_t)M * 512 * 2);
  float* fo   = (float*)alloc((size_t)M * 512 * 4);
  float* xcur = (float*)alloc((size_t)M * 513 * 4);
  bf16*  a2   = (bf16*)alloc((size_t)M * K2 * 2);
  bf16*  wqsb = (bf16*)alloc((size_t)512 * KD * 2);
  bf16*  wksb = (bf16*)alloc((size_t)512 * KD * 2);
  bf16*  wvsb = (bf16*)alloc((size_t)512 * KD * 2);
  bf16*  wqcb = (bf16*)alloc((size_t)512 * KD * 2);
  bf16*  wkcb = (bf16*)alloc((size_t)512 * KD * 2);
  bf16*  wvcb = (bf16*)alloc((size_t)512 * KD * 2);
  bf16*  wosb = (bf16*)alloc((size_t)512 * 512 * 2);
  bf16*  wocb = (bf16*)alloc((size_t)512 * 512 * 2);
  bf16*  w1b  = (bf16*)alloc((size_t)2048 * KD * 2);
  bf16*  w2b  = (bf16*)alloc((size_t)512 * K2 * 2);

  auto pack = [&](const float* s, bf16* d, int rows, int cols, int ldp) {
    size_t total = (size_t)rows * ldp;
    int grid = (int)((total + 255) / 256);
    if (grid > 8192) grid = 8192;
    pack_pad_bf16<<<grid, 256, 0, stream>>>(s, d, rows, cols, ldp);
  };
  auto gemm = [&](const bf16* A, int lda, const bf16* W, int ldw,
                  const float* bias, float* C, int N, int kIters) {
    gemm_bf16_f32<<<dim3(M / 64, N / 64), 128, 0, stream>>>(A, lda, W, ldw, bias, C, N, kIters);
  };

  // ---- pack inputs and weights to bf16 (padded) ----
  pack(X, xb, M, 513, KD);
  pack(ENC, encb, M, 513, KD);
  pack(Wq_s, wqsb, 512, 513, KD);
  pack(Wk_s, wksb, 512, 513, KD);
  pack(Wv_s, wvsb, 512, 513, KD);
  pack(Wq_c, wqcb, 512, 513, KD);
  pack(Wk_c, wkcb, 512, 513, KD);
  pack(Wv_c, wvcb, 512, 513, KD);
  pack(Wo_s, wosb, 512, 512, 512);
  pack(Wo_c, wocb, 512, 512, 512);
  pack(W1, w1b, 2048, 513, KD);
  pack(W2, w2b, 512, 2049, K2);

  // ---- self-attention ----
  gemm(xb, KD, wqsb, KD, bq_s, fo, 512, KD / 32);
  qk_pack<<<256, 256, 0, stream>>>(fo, qb, -1.0f);
  gemm(xb, KD, wksb, KD, bk_s, fo, 512, KD / 32);
  qk_pack<<<256, 256, 0, stream>>>(fo, kb, 1.0f);
  gemm(xb, KD, wvsb, KD, bv_s, fo, 512, KD / 32);
  v_pack<<<dim3(4, 64), 256, 0, stream>>>(fo, vtb);
  attention_kernel<<<dim3(64, T / 16), 32, 0, stream>>>(qb, kb, vtb, attn, T, S, 1);
  gemm(attn, 512, wosb, 512, bo_s, fo, 512, 512 / 32);
  residual_ln<<<M, 256, 0, stream>>>(X, fo, g1, be1, xcur, actb);

  // ---- cross-attention ----
  gemm(actb, KD, wqcb, KD, bq_c, fo, 512, KD / 32);
  qk_pack<<<256, 256, 0, stream>>>(fo, qb, -1.0f);
  gemm(encb, KD, wkcb, KD, bk_c, fo, 512, KD / 32);
  qk_pack<<<256, 256, 0, stream>>>(fo, kb, 1.0f);
  gemm(encb, KD, wvcb, KD, bv_c, fo, 512, KD / 32);
  v_pack<<<dim3(4, 64), 256, 0, stream>>>(fo, vtb);
  attention_kernel<<<dim3(64, T / 16), 32, 0, stream>>>(qb, kb, vtb, attn, T, S, 0);
  gemm(attn, 512, wocb, 512, bo_c, fo, 512, 512 / 32);
  residual_ln<<<M, 256, 0, stream>>>(xcur, fo, g2, be2, xcur, actb);

  // ---- FFN ----
  gemm_bf16_relu<<<dim3(M / 64, 2048 / 64), 128, 0, stream>>>(
      actb, KD, w1b, KD, b1, a2, K2, 1, KD / 32);
  ffn_time<<<M, 256, 0, stream>>>(a2);
  gemm(a2, K2, w2b, K2, b2, fo, 512, K2 / 32);
  residual_ln<<<M, 256, 0, stream>>>(xcur, fo, g3, be3, (float*)d_out, nullptr);
}
